// CoaT_61366492725765
// MI455X (gfx1250) — compile-verified, matrix operand order checked by reading, not run on hardware
//
#include <hip/hip_runtime.h>
#include <hip/hip_bf16.h>

typedef __bf16 bf16_t;
typedef __attribute__((ext_vector_type(16))) bf16_t v16bf;
typedef __attribute__((ext_vector_type(8)))  bf16_t v8bf;
typedef __attribute__((ext_vector_type(8)))  float  v8f;

#define BATCH 32
#define SEQ   785
#define SEQP  800
#define CDIM  512
#define HEADS 8
#define CHD   64
#define TC    1536
#define MROWS (BATCH*SEQ)   /* 25120, %32==0 */
#define MP    (BATCH*SEQP)  /* 25600 */

// ---------------- fp32 -> bf16 convert ----------------
__global__ void cvt_f32_bf16(const float* __restrict__ src, bf16_t* __restrict__ dst, int n) {
  int i = (blockIdx.x * blockDim.x + threadIdx.x) * 4;
  if (i + 3 < n) {
    float4 f = *(const float4*)(src + i);
    dst[i+0] = (bf16_t)f.x; dst[i+1] = (bf16_t)f.y;
    dst[i+2] = (bf16_t)f.z; dst[i+3] = (bf16_t)f.w;
  } else {
    for (; i < n; ++i) dst[i] = (bf16_t)src[i];
  }
}

// ---------------- WMMA helpers ----------------
__device__ __forceinline__ v16bf ld_tile16(const bf16_t* p) {
  v8bf c0 = *(const v8bf*)(p);
  v8bf c1 = *(const v8bf*)(p + 16);
  return __builtin_shufflevector(c0, c1, 0,1,2,3,4,5,6,7,8,9,10,11,12,13,14,15);
}

#define WMMA_BF16(a, b, c) \
  __builtin_amdgcn_wmma_f32_16x16x32_bf16(false, (a), false, (b), (short)0, (c), false, false)

// One wave computes a 32(M) x 64(N) fp32 tile. A: MxK row-major, Bt: NxK row-major.
// Compile-time strides/K so all row offsets fold into instruction offsets; pointers
// advance by 32 elements per K-step. Next step's 12 loads issue before current 8 WMMAs;
// full unroll lets SSA renaming eliminate the buffer rotation.
template<int LDA, int LDB, int K>
__device__ __forceinline__ void gemm32x64(const bf16_t* __restrict__ A,
                                          const bf16_t* __restrict__ Bt,
                                          v8f c[8]) {
  const int lane = threadIdx.x & 31;
  const int row  = lane & 15;
  const int half = lane >> 4;
  const bf16_t* ap = A  + row * LDA + half * 8;
  const bf16_t* bp = Bt + row * LDB + half * 8;

  v16bf a0 = ld_tile16(ap);
  v16bf a1 = ld_tile16(ap + 16 * LDA);
  v16bf b0 = ld_tile16(bp);
  v16bf b1 = ld_tile16(bp + 16 * LDB);
  v16bf b2 = ld_tile16(bp + 32 * LDB);
  v16bf b3 = ld_tile16(bp + 48 * LDB);

  constexpr int nsteps = K >> 5;
#pragma unroll
  for (int s = 0; s < nsteps - 1; ++s) {
    ap += 32; bp += 32;
    // prefetch next K-step
    v16bf na0 = ld_tile16(ap);
    v16bf na1 = ld_tile16(ap + 16 * LDA);
    v16bf nb0 = ld_tile16(bp);
    v16bf nb1 = ld_tile16(bp + 16 * LDB);
    v16bf nb2 = ld_tile16(bp + 32 * LDB);
    v16bf nb3 = ld_tile16(bp + 48 * LDB);
    // consume current K-step
    c[0] = WMMA_BF16(a0, b0, c[0]);
    c[1] = WMMA_BF16(a0, b1, c[1]);
    c[2] = WMMA_BF16(a0, b2, c[2]);
    c[3] = WMMA_BF16(a0, b3, c[3]);
    c[4] = WMMA_BF16(a1, b0, c[4]);
    c[5] = WMMA_BF16(a1, b1, c[5]);
    c[6] = WMMA_BF16(a1, b2, c[6]);
    c[7] = WMMA_BF16(a1, b3, c[7]);
    a0 = na0; a1 = na1; b0 = nb0; b1 = nb1; b2 = nb2; b3 = nb3;
  }
  c[0] = WMMA_BF16(a0, b0, c[0]);
  c[1] = WMMA_BF16(a0, b1, c[1]);
  c[2] = WMMA_BF16(a0, b2, c[2]);
  c[3] = WMMA_BF16(a0, b3, c[3]);
  c[4] = WMMA_BF16(a1, b0, c[4]);
  c[5] = WMMA_BF16(a1, b1, c[5]);
  c[6] = WMMA_BF16(a1, b2, c[6]);
  c[7] = WMMA_BF16(a1, b3, c[7]);
}

// ---------------- K1: qkv = x @ qkv_w.T + b  (25120 x 1536, K=512) ----------------
__global__ void __launch_bounds__(256, 1)
gemm_qkv(const bf16_t* __restrict__ xb, const bf16_t* __restrict__ wb,
         const float* __restrict__ bias, float* __restrict__ out) {
  int w = blockIdx.x * 8 + (threadIdx.x >> 5);
  if (w >= (MROWS/32) * (TC/64)) return;
  int mt = w / (TC/64), nt = w % (TC/64);
  v8f c[8] = {};
  gemm32x64<CDIM, CDIM, CDIM>(xb + (size_t)(mt*32) * CDIM, wb + (size_t)(nt*64) * CDIM, c);
  int lane = threadIdx.x & 31, cl = lane & 15, half = lane >> 4;
#pragma unroll
  for (int mh = 0; mh < 2; ++mh) {
    int rbase = mt*32 + mh*16 + half*8;
#pragma unroll
    for (int j = 0; j < 4; ++j) {
      int col = nt*64 + j*16 + cl;
      float bv = bias[col];
#pragma unroll
      for (int v = 0; v < 8; ++v)
        out[(size_t)(rbase + v) * TC + col] = c[mh*4 + j][v] + bv;
    }
  }
}

// ---------------- K2: column softmax of k over N + pack transposed bf16 ----------------
__global__ void softmax_pack(const float* __restrict__ qkv, bf16_t* __restrict__ keT,
                             bf16_t* __restrict__ vT, bf16_t* __restrict__ qb) {
  int b = blockIdx.x, t = threadIdx.x;              // 512 threads = one channel each
  const float* base = qkv + (size_t)b * SEQ * TC;
  float m = -__builtin_inff();
  for (int n = 0; n < SEQ; ++n) m = fmaxf(m, base[(size_t)n*TC + CDIM + t]);
  float s = 0.f;
  for (int n = 0; n < SEQ; ++n) s += __expf(base[(size_t)n*TC + CDIM + t] - m);
  float inv = 1.f / s;
  bf16_t* kc = keT + ((size_t)b * CDIM + t) * SEQP;
  bf16_t* vc = vT  + ((size_t)b * CDIM + t) * SEQP;
  for (int n = 0; n < SEQ; ++n) {
    kc[n] = (bf16_t)(__expf(base[(size_t)n*TC + CDIM + t] - m) * inv);
    vc[n] = (bf16_t)(base[(size_t)n*TC + 2*CDIM + t]);
    qb[((size_t)b*SEQP + n) * CDIM + t] = (bf16_t)(base[(size_t)n*TC + t]);
  }
  for (int n = SEQ; n < SEQP; ++n) {
    kc[n] = (bf16_t)0.f; vc[n] = (bf16_t)0.f;
    qb[((size_t)b*SEQP + n) * CDIM + t] = (bf16_t)0.f;
  }
}

// ---------------- K3: kvT[bh][d][c] = sum_n v[n,d]*k_sm[n,c]  (64x64, K=800) ----------------
__global__ void __launch_bounds__(256, 1)
gemm_kv(const bf16_t* __restrict__ vT, const bf16_t* __restrict__ keT,
        bf16_t* __restrict__ kvT) {
  int w = blockIdx.x * 8 + (threadIdx.x >> 5);      // 0..511
  if (w >= BATCH*HEADS*2) return;
  int bh = w >> 1, mt = w & 1;                      // bh*64 == b*512 + hd*64
  v8f c[8] = {};
  gemm32x64<SEQP, SEQP, SEQP>(vT + ((size_t)bh*64 + mt*32) * SEQP,
                              keT + (size_t)bh*64 * SEQP, c);
  int lane = threadIdx.x & 31, cl = lane & 15, half = lane >> 4;
  bf16_t* outp = kvT + (size_t)bh * 64 * 64;
#pragma unroll
  for (int mh = 0; mh < 2; ++mh) {
    int rbase = mt*32 + mh*16 + half*8;
#pragma unroll
    for (int j = 0; j < 4; ++j)
#pragma unroll
      for (int v = 0; v < 8; ++v)
        outp[(size_t)(rbase + v) * 64 + j*16 + cl] = (bf16_t)c[mh*4 + j][v];
  }
}

// ---------------- K4: attn = scale * (q @ kv)  (800 rows per (b,h), K=64) ----------------
__global__ void __launch_bounds__(256, 1)
gemm_fa(const bf16_t* __restrict__ qb, const bf16_t* __restrict__ kvT,
        float* __restrict__ attn) {
  int w = blockIdx.x * 8 + (threadIdx.x >> 5);      // 0..6399
  if (w >= BATCH*HEADS*(SEQP/32)) return;
  int bh = w / (SEQP/32), mt = w % (SEQP/32);
  int b = bh >> 3, hd = bh & 7;
  v8f c[8] = {};
  gemm32x64<CDIM, 64, 64>(qb + ((size_t)b*SEQP + mt*32) * CDIM + hd*CHD,
                          kvT + (size_t)bh * 64 * 64, c);
  int lane = threadIdx.x & 31, cl = lane & 15, half = lane >> 4;
  const float scale = 0.125f;                       // Ch^-0.5
#pragma unroll
  for (int mh = 0; mh < 2; ++mh) {
    int rbase = mt*32 + mh*16 + half*8;
#pragma unroll
    for (int j = 0; j < 4; ++j)
#pragma unroll
      for (int v = 0; v < 8; ++v)
        attn[((size_t)b*SEQP + rbase + v) * CDIM + hd*CHD + j*16 + cl] = scale * c[mh*4 + j][v];
  }
}

// ---------------- K5: depthwise conv (3/5/7) fused with EV=q*conv, += into attn ----------------
__global__ void conv_ev(const float* __restrict__ qkv,
                        const float* __restrict__ w3, const float* __restrict__ b3,
                        const float* __restrict__ w5, const float* __restrict__ b5,
                        const float* __restrict__ w7, const float* __restrict__ b7,
                        float* __restrict__ attn) {
  __shared__ float in_lds[7 * 28 * 64];
  __shared__ float w_lds[64 * 49];
  __shared__ float bias_lds[64];
  int g = blockIdx.x, y = blockIdx.y, b = blockIdx.z;   // 8 ch-groups, 28 rows, 32 batch
  int tid = threadIdx.x;                                 // 256 threads
  int ks, wofs; const float* W; const float* Bv;
  if (g < 2)      { ks = 3; W = w3; Bv = b3; wofs = g*64;       }
  else if (g < 5) { ks = 5; W = w5; Bv = b5; wofs = g*64 - 128; }
  else            { ks = 7; W = w7; Bv = b7; wofs = g*64 - 320; }
  int r = ks >> 1, rows = 2*r + 1, k2 = ks*ks;
  for (int i = tid; i < 64*k2; i += 256) w_lds[i] = W[(size_t)wofs * k2 + i];
  for (int i = tid; i < 64; i += 256)    bias_lds[i] = Bv[wofs + i];
  int tot = rows * 28 * 64;
  for (int i = tid; i < tot; i += 256) {
    int cc = i & 63, x = (i >> 6) % 28, rr = i / (64*28);
    int yy = y - r + rr;
    float vv = 0.f;
    if (yy >= 0 && yy < 28)
      vv = qkv[((size_t)b*SEQ + 1 + yy*28 + x) * TC + 2*CDIM + g*64 + cc];
    in_lds[(rr*28 + x)*64 + cc] = vv;
  }
  __syncthreads();
#pragma unroll
  for (int j = 0; j < 7; ++j) {
    int e = tid + j*256;          // 0..1791 = 28 px * 64 ch
    int cc = e & 63, x = e >> 6;
    float acc = bias_lds[cc];
    const float* wp = &w_lds[cc * k2];
    for (int dy = 0; dy < rows; ++dy)
      for (int dx = -r; dx <= r; ++dx) {
        int xx = x + dx;
        if (xx >= 0 && xx < 28)
          acc += in_lds[(dy*28 + xx)*64 + cc] * wp[dy*ks + dx + r];
      }
    int n = 1 + y*28 + x, col = g*64 + cc;
    float qv = qkv[((size_t)b*SEQ + n) * TC + col];
    attn[((size_t)b*SEQP + n) * CDIM + col] += qv * acc;
  }
}

// ---------------- K7: out = attn @ proj_w.T + b  (mask 800->785 rows) ----------------
__global__ void __launch_bounds__(256, 1)
gemm_proj(const bf16_t* __restrict__ ab, const bf16_t* __restrict__ wb,
          const float* __restrict__ bias, float* __restrict__ out) {
  int w = blockIdx.x * 8 + (threadIdx.x >> 5);      // 0..6399
  if (w >= (MP/32) * (CDIM/64)) return;
  int mt = w >> 3, nt = w & 7;
  v8f c[8] = {};
  gemm32x64<CDIM, CDIM, CDIM>(ab + (size_t)(mt*32) * CDIM, wb + (size_t)(nt*64) * CDIM, c);
  int lane = threadIdx.x & 31, cl = lane & 15, half = lane >> 4;
#pragma unroll
  for (int mh = 0; mh < 2; ++mh) {
    int rbase = mt*32 + mh*16 + half*8;
#pragma unroll
    for (int j = 0; j < 4; ++j) {
      int col = nt*64 + j*16 + cl;
      float bv = bias[col];
#pragma unroll
      for (int v = 0; v < 8; ++v) {
        int m = rbase + v, b = m / SEQP, n = m % SEQP;
        if (n < SEQ)
          out[((size_t)b*SEQ + n) * CDIM + col] = c[mh*4 + j][v] + bv;
      }
    }
  }
}

// ---------------- host launcher ----------------
extern "C" void kernel_launch(void* const* d_in, const int* in_sizes, int n_in,
                              void* d_out, int out_size, void* d_ws, size_t ws_size,
                              hipStream_t stream) {
  (void)in_sizes; (void)n_in; (void)out_size; (void)ws_size;
  const float* x      = (const float*)d_in[0];
  const float* qkv_w  = (const float*)d_in[1];
  const float* qkv_b  = (const float*)d_in[2];
  const float* proj_w = (const float*)d_in[3];
  const float* proj_b = (const float*)d_in[4];
  const float* w3 = (const float*)d_in[5];
  const float* b3 = (const float*)d_in[6];
  const float* w5 = (const float*)d_in[7];
  const float* b5 = (const float*)d_in[8];
  const float* w7 = (const float*)d_in[9];
  const float* b7 = (const float*)d_in[10];
  float* out = (float*)d_out;

  char* p = (char*)d_ws;
  auto carve = [&](size_t bytes) { void* r = (void*)p; p += (bytes + 255) & ~(size_t)255; return r; };
  bf16_t* x_bf   = (bf16_t*)carve((size_t)MROWS * CDIM * 2);
  bf16_t* wqkv_bf= (bf16_t*)carve((size_t)TC * CDIM * 2);
  bf16_t* wprj_bf= (bf16_t*)carve((size_t)CDIM * CDIM * 2);
  float*  qkv    = (float*) carve((size_t)MROWS * TC * 4);
  bf16_t* keT    = (bf16_t*)carve((size_t)BATCH * CDIM * SEQP * 2);
  bf16_t* vT     = (bf16_t*)carve((size_t)BATCH * CDIM * SEQP * 2);
  bf16_t* q_bf   = (bf16_t*)carve((size_t)MP * CDIM * 2);
  bf16_t* kvT    = (bf16_t*)carve((size_t)BATCH * HEADS * 64 * 64 * 2);
  float*  attn   = (float*) carve((size_t)MP * CDIM * 4);
  bf16_t* attn_bf= (bf16_t*)carve((size_t)MP * CDIM * 2);

  auto cvt = [&](const float* s, bf16_t* d, int n) {
    int blks = (n/4 + 255) / 256;
    cvt_f32_bf16<<<blks, 256, 0, stream>>>(s, d, n);
  };
  cvt(x,      x_bf,    MROWS * CDIM);
  cvt(qkv_w,  wqkv_bf, TC * CDIM);
  cvt(proj_w, wprj_bf, CDIM * CDIM);

  gemm_qkv<<<(MROWS/32)*(TC/64)/8, 256, 0, stream>>>(x_bf, wqkv_bf, qkv_b, qkv);
  softmax_pack<<<BATCH, 512, 0, stream>>>(qkv, keT, vT, q_bf);
  gemm_kv<<<(BATCH*HEADS*2)/8, 256, 0, stream>>>(vT, keT, kvT);
  gemm_fa<<<(BATCH*HEADS*(SEQP/32))/8, 256, 0, stream>>>(q_bf, kvT, attn);
  conv_ev<<<dim3(8, 28, BATCH), 256, 0, stream>>>(qkv, w3, b3, w5, b5, w7, b7, attn);
  cvt((const float*)attn, attn_bf, MP * CDIM);
  gemm_proj<<<((MP/32)*(CDIM/64))/8, 256, 0, stream>>>(attn_bf, wprj_bf, proj_b, out);
}
